// arDCA_paths_4776003633350
// MI455X (gfx1250) — matrix operand order repaired and checked by reading, not compiled, not run on hardware
//
#include <hip/hip_runtime.h>
#include <hip/hip_bf16.h>

// Problem constants (from reference)
#define LL   200
#define QQ   21
#define MM   1024
#define START 133            // 2*L//3
#define SS   67              // L - START
#define RROWS (SS*QQ)        // 1407
#define RPAD  1408           // 88 tiles of 16
#define RTILES 88
#define KK   (LL*QQ)         // 4200
#define KPAD 4224            // 132 chunks of 32
#define KCHUNKS 132
#define NTILES (MM/16)       // 64
#define REG_H 0.01f
#define REG_J 0.01f

typedef __attribute__((ext_vector_type(16))) _Float16 v16h;
typedef __attribute__((ext_vector_type(8)))  float    v8f;

// ---- workspace layout (bytes) ----
#define A_BYTES   ((size_t)RTILES * KCHUNKS * 32 * 16 * 2)   // 11,894,784
#define B_BYTES   ((size_t)MM * KPAD * 2)                    //  8,650,752
#define C_BYTES   ((size_t)RPAD * MM * 4)                    //  5,767,168
#define OFF_A     ((size_t)0)
#define OFF_B     (OFF_A + A_BYTES)
#define OFF_C     (OFF_B + B_BYTES)
#define OFF_PREGJ (OFF_C + C_BYTES)
#define OFF_PJFIJ (OFF_PREGJ + 4096)
#define OFF_PLOGZ (OFF_PJFIJ + 4096)
#define JRED_BLOCKS 1024
#define LOGZ_BLOCKS 268      // (M*S)/256 = 68608/256

// ---------------------------------------------------------------------------
// Build A operand: masked J (j < i), f16, pre-swizzled to a K-contiguous (k = 16*h + s) A-matrix
// 16x32 fragment layout so each lane loads 16 contiguous halves per chunk.
//   storage index: (((T*KCHUNKS + kc)*32 + lane)*16 + d)
//   lane 0-15 : m=lane,    VGPR0-7: K=0..15
//   lane 16-31: m=lane-16, VGPR0-7: K=16..31
// ---------------------------------------------------------------------------
__global__ void build_A_kernel(const float* __restrict__ J, _Float16* __restrict__ A) {
    size_t t = (size_t)blockIdx.x * blockDim.x + threadIdx.x;   // one half per thread
    int d  = (int)(t & 15);
    int l  = (int)((t >> 4) & 31);
    int kc = (int)((t >> 9) % KCHUNKS);
    int T  = (int)(t / ((size_t)512 * KCHUNKS));
    if (T >= RTILES) return;

    int m  = l & 15;
    int hi = l >> 4;                 // lane group
    int vg = d >> 1;
    int hb = d & 1;
    int Klocal = 2 * vg + hb + 16 * hi;
    int kglob = kc * 32 + Klocal;
    int r = T * 16 + m;

    float val = 0.0f;
    if (r < RROWS && kglob < KK) {
        int i = START + r / QQ;
        int a = r % QQ;
        int j = kglob / QQ;
        int b = kglob % QQ;
        if (j < i) {
            val = J[(((size_t)i * QQ + a) * LL + j) * QQ + b];
        }
    }
    A[t] = (_Float16)val;
}

// ---------------------------------------------------------------------------
// Build B operand: one-hot(X_tokens), f16, column-major (K contiguous per n).
// Here the B fragment is K-contiguous per lane (k = 16*h + s, as in A), so this loads directly.
// ---------------------------------------------------------------------------
__global__ void build_B_kernel(const int* __restrict__ X, _Float16* __restrict__ B) {
    size_t t = (size_t)blockIdx.x * blockDim.x + threadIdx.x;
    int k = (int)(t % KPAD);
    int n = (int)(t / KPAD);
    if (n >= MM) return;
    float v = 0.0f;
    if (k < KK) {
        int j = k / QQ;
        int b = k % QQ;
        v = (X[n * LL + j] == b) ? 1.0f : 0.0f;
    }
    B[t] = (_Float16)v;
}

// ---------------------------------------------------------------------------
// WMMA GEMM with 2x2 register blocking: each wave owns a 32x32 output block
// (2 row tiles x 2 col tiles), 4 WMMAs per 32-wide K chunk from 2+2 fragment
// loads -> 2x arithmetic intensity vs 1x1 tiling.  44*32 = 1408 waves.
// ---------------------------------------------------------------------------
__global__ void gemm_wmma_kernel(const _Float16* __restrict__ A,
                                 const _Float16* __restrict__ B,
                                 float* __restrict__ C) {
    int wave = threadIdx.x >> 5;
    int lane = threadIdx.x & 31;
    int tile = blockIdx.x * 8 + wave;        // grid sized exactly: 176*8 = 1408
    int tR = tile >> 5;                      // 0..43 (pairs of row tiles)
    int tN = tile & 31;                      // 0..31 (pairs of col tiles)
    int tr0 = tR * 2;
    int tn0 = tN * 2;

    const _Float16* ap0 = A + (((size_t)tr0 * KCHUNKS) * 32 + lane) * 16;
    const _Float16* ap1 = ap0 + (size_t)KCHUNKS * 512;          // next row tile
    const _Float16* bp0 = B + (size_t)(tn0 * 16 + (lane & 15)) * KPAD + (lane >> 4) * 16;
    const _Float16* bp1 = bp0 + (size_t)16 * KPAD;              // next col tile

    v8f acc00 = {}, acc01 = {}, acc10 = {}, acc11 = {};
    #pragma unroll 2
    for (int kc = 0; kc < KCHUNKS; ++kc) {
        v16h a0 = *(const v16h*)(ap0);
        v16h a1 = *(const v16h*)(ap1);
        v16h b0 = *(const v16h*)(bp0);
        v16h b1 = *(const v16h*)(bp1);
        ap0 += 512; ap1 += 512;              // next chunk: 32 lanes * 16 halves
        bp0 += 32;  bp1 += 32;               // next 32 K values in this column
        acc00 = __builtin_amdgcn_wmma_f32_16x16x32_f16(
                    false, a0, false, b0, (short)0, acc00, false, false);
        acc01 = __builtin_amdgcn_wmma_f32_16x16x32_f16(
                    false, a0, false, b1, (short)0, acc01, false, false);
        acc10 = __builtin_amdgcn_wmma_f32_16x16x32_f16(
                    false, a1, false, b0, (short)0, acc10, false, false);
        acc11 = __builtin_amdgcn_wmma_f32_16x16x32_f16(
                    false, a1, false, b1, (short)0, acc11, false, false);
    }

    // C fragment layout: VGPR v -> M = v + 8*(lane>=16), N = lane%16
    int moff = (lane >> 4) ? 8 : 0;
    int ncol = lane & 15;
    #pragma unroll
    for (int v = 0; v < 8; ++v) {
        size_t r0 = (size_t)(tr0 * 16 + moff + v);
        size_t r1 = (size_t)((tr0 + 1) * 16 + moff + v);
        C[r0 * MM + tn0 * 16 + ncol]        = acc00[v];
        C[r0 * MM + (tn0 + 1) * 16 + ncol]  = acc01[v];
        C[r1 * MM + tn0 * 16 + ncol]        = acc10[v];
        C[r1 * MM + (tn0 + 1) * 16 + ncol]  = acc11[v];
    }
}

// ---------------------------------------------------------------------------
// Deterministic block reduce helper (blockDim.x == 256 everywhere below).
// ---------------------------------------------------------------------------
__device__ __forceinline__ float block_reduce256(float v, float* sm) {
    int t = threadIdx.x;
    sm[t] = v;
    __syncthreads();
    for (int s = 128; s > 0; s >>= 1) {
        if (t < s) sm[t] += sm[t + s];
        __syncthreads();
    }
    float r = sm[0];
    __syncthreads();
    return r;
}

// ---------------------------------------------------------------------------
// Single fused streaming pass over J / fij: reg_J = sum J^2 (all elements),
// jfij = sum over (i>=START, j<i) of J*fij.  Fixed grid -> deterministic.
// This is the HBM-roofline part: ~141 MB read once.
// ---------------------------------------------------------------------------
__global__ void jreduce_kernel(const float* __restrict__ J,
                               const float* __restrict__ fij,
                               float* __restrict__ pRegJ,
                               float* __restrict__ pJfij) {
    __shared__ float sm[256];
    const size_t N = (size_t)LL * QQ * LL * QQ;   // 17,640,000
    size_t stride = (size_t)gridDim.x * blockDim.x;
    float regj = 0.0f, jf = 0.0f;
    for (size_t idx = (size_t)blockIdx.x * blockDim.x + threadIdx.x; idx < N; idx += stride) {
        float jv = J[idx];
        regj += jv * jv;
        int i = (int)(idx / (QQ * LL * QQ));          // / 88200
        int rem = (int)(idx % (QQ * LL * QQ));
        int j = (rem % (LL * QQ)) / QQ;
        if (i >= START && j < i) jf += jv * fij[idx];
    }
    float r0 = block_reduce256(regj, sm);
    float r1 = block_reduce256(jf, sm);
    if (threadIdx.x == 0) { pRegJ[blockIdx.x] = r0; pJfij[blockIdx.x] = r1; }
}

// ---------------------------------------------------------------------------
// logZ per (n, site) + weighted partial sums. 268 blocks * 256 = 68608 pairs.
// ---------------------------------------------------------------------------
__global__ void logz_kernel(const float* __restrict__ C,
                            const float* __restrict__ h,
                            const float* __restrict__ weights,
                            float* __restrict__ pLogz) {
    __shared__ float sm[256];
    int p = blockIdx.x * 256 + threadIdx.x;   // p = s*1024 + n
    int n = p & (MM - 1);
    int s = p >> 10;                          // 0..66
    float acc = 0.0f;
    if (s < SS) {
        const float* hrow = h + (size_t)(START + s) * QQ;
        const float* crow = C + (size_t)(s * QQ) * MM + n;
        float mx = -3.4e38f;
        float lg[QQ];
        #pragma unroll
        for (int a = 0; a < QQ; ++a) {
            float v = crow[(size_t)a * MM] + hrow[a];
            lg[a] = v;
            mx = fmaxf(mx, v);
        }
        float se = 0.0f;
        #pragma unroll
        for (int a = 0; a < QQ; ++a) se += __expf(lg[a] - mx);
        float logZ = mx + __logf(se);
        acc = logZ * weights[n];
    }
    float r = block_reduce256(acc, sm);
    if (threadIdx.x == 0) pLogz[blockIdx.x] = r;
}

// ---------------------------------------------------------------------------
// Final combine: small reductions + assemble (loss, nll). One block.
// ---------------------------------------------------------------------------
__global__ void finalize_kernel(const float* __restrict__ weights,
                                const float* __restrict__ fi,
                                const float* __restrict__ h,
                                const float* __restrict__ pRegJ,
                                const float* __restrict__ pJfij,
                                const float* __restrict__ pLogz,
                                float* __restrict__ out) {
    __shared__ float sm[256];
    int t = threadIdx.x;

    float v = 0.0f;
    for (int k = t; k < MM; k += 256) v += weights[k];
    float sumw = block_reduce256(v, sm);

    v = 0.0f;
    for (int k = t; k < RROWS; k += 256) {
        int idx = START * QQ + k;
        v += fi[idx] * h[idx];
    }
    float fih = block_reduce256(v, sm);

    v = 0.0f;
    for (int k = t; k < LL * QQ; k += 256) v += h[k] * h[k];
    float regh = block_reduce256(v, sm);

    v = 0.0f;
    for (int k = t; k < JRED_BLOCKS; k += 256) v += pRegJ[k];
    float regJ = block_reduce256(v, sm);

    v = 0.0f;
    for (int k = t; k < JRED_BLOCKS; k += 256) v += pJfij[k];
    float jfij = block_reduce256(v, sm);

    v = 0.0f;
    for (int k = t; k < LOGZ_BLOCKS; k += 256) v += pLogz[k];
    float logzw = block_reduce256(v, sm);

    if (t == 0) {
        float energy = fih + jfij;
        float nll = logzw / sumw - energy;      // -(energy - sum logZ * w)
        float loss = nll + REG_H * regh + REG_J * regJ;
        out[0] = loss;
        out[1] = nll;
    }
}

// ---------------------------------------------------------------------------
extern "C" void kernel_launch(void* const* d_in, const int* in_sizes, int n_in,
                              void* d_out, int out_size, void* d_ws, size_t ws_size,
                              hipStream_t stream) {
    const int*   X       = (const int*)d_in[0];     // (M, L)
    const float* weights = (const float*)d_in[1];   // (M,)
    const float* fi      = (const float*)d_in[2];   // (L, Q)
    const float* fij     = (const float*)d_in[3];   // (L, Q, L, Q)
    const float* h       = (const float*)d_in[4];   // (L, Q)
    const float* J       = (const float*)d_in[5];   // (L, Q, L, Q)
    float* out = (float*)d_out;

    char* ws = (char*)d_ws;
    _Float16* A     = (_Float16*)(ws + OFF_A);
    _Float16* B     = (_Float16*)(ws + OFF_B);
    float*    C     = (float*)   (ws + OFF_C);
    float*    pRegJ = (float*)   (ws + OFF_PREGJ);
    float*    pJfij = (float*)   (ws + OFF_PJFIJ);
    float*    pLogz = (float*)   (ws + OFF_PLOGZ);

    // 1) Pack operands (A pre-swizzled into WMMA fragment layout, B column-major one-hot)
    {
        size_t total = (size_t)RTILES * KCHUNKS * 32 * 16;          // 5,947,392
        build_A_kernel<<<(unsigned)(total / 256), 256, 0, stream>>>(J, A);
    }
    {
        size_t total = (size_t)MM * KPAD;                           // 4,325,376
        build_B_kernel<<<(unsigned)(total / 256), 256, 0, stream>>>(X, B);
    }

    // 2) WMMA GEMM with 2x2 register blocking: 44*32 tiles of 32x32, 8 waves/block
    gemm_wmma_kernel<<<(44 * 32) / 8, 256, 0, stream>>>(A, B, C);

    // 3) Streaming J/fij reductions (HBM-bound part, single fused pass)
    jreduce_kernel<<<JRED_BLOCKS, 256, 0, stream>>>(J, fij, pRegJ, pJfij);

    // 4) logsumexp + weighted partials
    logz_kernel<<<LOGZ_BLOCKS, 256, 0, stream>>>(C, h, weights, pLogz);

    // 5) Deterministic final combine
    finalize_kernel<<<1, 256, 0, stream>>>(weights, fi, h, pRegJ, pJfij, pLogz, out);
}
